// NLB_Head_AV_77481210020382
// MI455X (gfx1250) — compile-verified, matrix operand order checked by reading, not run on hardware
//
#include <hip/hip_runtime.h>
#include <hip/hip_bf16.h>

// ---------------------------------------------------------------------------
// NLB head, fused for MI455X (gfx1250, wave32).
//   scores = (theta @ W_phi) . lfb   -> streaming dot, online softmax
//   ctx    = W_gi @ (sum attn*lfb)   -> weighted streaming sum, tiny GEMM
// Streaming pass is HBM-bound: 512 MB @ 23.3 TB/s ~ 22 us (single pass via
// online softmax; a two-pass design would be 44 us since lfb > L2).
// Dense M=16 layers use v_wmma_f32_16x16x32_bf16.
// lfb loads are non-temporal (streamed once, 2.7x L2 size) and software-
// pipelined (next row's 4x b128 in flight during dot/reduce/axpy).
// ---------------------------------------------------------------------------

#define D 512
#define B 16
#define T 16384
#define NCHUNK 32
#define CHUNK (T / NCHUNK)   // 512 timesteps per workgroup
#define WITER (CHUNK / 8)    // 64 timesteps per wave
#define REC 520              // partial record: 512 acc + m + l (+pad)
#define SQRT_D 22.62741699796952f
#define LN_EPS 1e-5f

typedef __attribute__((ext_vector_type(16))) __bf16 v16bf;
typedef __attribute__((ext_vector_type(8)))  float  v8f;
typedef __attribute__((ext_vector_type(4)))  float  f4;

// ---------------- WMMA helpers (ISA 7.12.2 layouts) ----------------

static __device__ inline v8f wmma_bf16(v16bf a, v16bf b, v8f c) {
  return __builtin_amdgcn_wmma_f32_16x16x32_bf16(false, a, false, b,
                                                 (short)0, c, false, false);
}

static __device__ inline v8f zero_v8f() {
  v8f c;
#pragma unroll
  for (int i = 0; i < 8; ++i) c[i] = 0.0f;
  return c;
}

// A fragment, 16x32, from row-major f32 [16 x ld], rows 0..15, cols k0..k0+31.
// Lane L: m = L&15, kb = (L>>4)*8; holds K = {kb..kb+7, kb+16..kb+23}.
static __device__ inline v16bf load_a_rm(const float* __restrict__ src, int ld,
                                         int k0, int lane) {
  const int m  = lane & 15;
  const int kb = (lane >> 4) << 3;
  const float* p = src + (size_t)m * ld + k0 + kb;
  v16bf a;
#pragma unroll
  for (int i = 0; i < 8; ++i) {
    a[i]     = (__bf16)p[i];
    a[i + 8] = (__bf16)p[16 + i];
  }
  return a;
}

// B fragment, 32x16, for D = A @ W^T: B(k,n) = W[(n0+n)*ld + k]  (contiguous k)
static __device__ inline v16bf load_b_wT(const float* __restrict__ w, int ld,
                                         int k0, int n0, int lane) {
  const int n  = lane & 15;
  const int kb = (lane >> 4) << 3;
  const float* p = w + (size_t)(n0 + n) * ld + k0 + kb;
  v16bf b;
#pragma unroll
  for (int i = 0; i < 8; ++i) {
    b[i]     = (__bf16)p[i];
    b[i + 8] = (__bf16)p[16 + i];
  }
  return b;
}

// B fragment for D = A @ W (no transpose): B(k,n) = W[k*ld + n0+n] (strided).
// Only used for the one-time 1 MB read of w_phi (~43 ns of HBM traffic).
static __device__ inline v16bf load_b_w(const float* __restrict__ w, int ld,
                                        int k0, int n0, int lane) {
  const int n  = lane & 15;
  const int kb = (lane >> 4) << 3;
  const float* p = w + (size_t)(k0 + kb) * ld + n0 + n;
  v16bf b;
#pragma unroll
  for (int i = 0; i < 8; ++i) {
    b[i]     = (__bf16)p[(size_t)i * ld];
    b[i + 8] = (__bf16)p[(size_t)(16 + i) * ld];
  }
  return b;
}

static __device__ inline float wave_sum(float p) {
#pragma unroll
  for (int off = 16; off > 0; off >>= 1) p += __shfl_xor(p, off, 32);
  return p;
}

// ---------------- Kernel 1: theta = x@Wth^T + bth ; u = sqrt(D)*theta@Wphi ---

__global__ void nlb_head(const float* __restrict__ x,
                         const float* __restrict__ w_theta,
                         const float* __restrict__ b_theta,
                         const float* __restrict__ w_phi,
                         float* __restrict__ u) {
  __shared__ float th[B * D];
  const int lane = threadIdx.x & 31;
  const int wave = threadIdx.x >> 5;   // 16 waves
  const int rb   = (lane >> 4) << 3;
  const int nlo  = lane & 15;

  // Stage A: theta[16x512] -> LDS
  for (int tile = wave; tile < D / 16; tile += 16) {
    const int n0 = tile * 16;
    v8f c = zero_v8f();
    for (int k0 = 0; k0 < D; k0 += 32) {
      v16bf a = load_a_rm(x, D, k0, lane);
      v16bf b = load_b_wT(w_theta, D, k0, n0, lane);
      c = wmma_bf16(a, b, c);
    }
    const float bias = b_theta[n0 + nlo];
#pragma unroll
    for (int r = 0; r < 8; ++r) th[(rb + r) * D + n0 + nlo] = c[r] + bias;
  }
  __syncthreads();

  // Stage B: u = sqrt(D) * theta @ w_phi   (b_phi dropped: softmax-invariant)
  for (int tile = wave; tile < D / 16; tile += 16) {
    const int n0 = tile * 16;
    v8f c = zero_v8f();
    for (int k0 = 0; k0 < D; k0 += 32) {
      v16bf a = load_a_rm(th, D, k0, lane);
      v16bf b = load_b_w(w_phi, D, k0, n0, lane);
      c = wmma_bf16(a, b, c);
    }
#pragma unroll
    for (int r = 0; r < 8; ++r)
      u[(size_t)(rb + r) * D + n0 + nlo] = c[r] * SQRT_D;
  }
}

// ---------------- Kernel 2: streaming online-softmax weighted sum -----------

static __device__ inline f4 f4_axpy(f4 c, float e, f4 a) {
  c.x += e * a.x; c.y += e * a.y; c.z += e * a.z; c.w += e * a.w; return c;
}
static __device__ inline f4 f4_resc(f4 c, float r, f4 a) {
  c.x = c.x * r + a.x; c.y = c.y * r + a.y;
  c.z = c.z * r + a.z; c.w = c.w * r + a.w; return c;
}
static __device__ inline f4 ntld(const f4* p) {
  return __builtin_nontemporal_load(p);
}

__global__ void nlb_stream(const float* __restrict__ lfb,
                           const float* __restrict__ u,
                           float* __restrict__ part) {
  __shared__ float sm_acc[8][D];
  __shared__ float sm_m[8];
  __shared__ float sm_l[8];

  const int b     = blockIdx.x >> 5;       // NCHUNK == 32
  const int chunk = blockIdx.x & (NCHUNK - 1);
  const int tid   = threadIdx.x;           // 256 threads, 8 waves
  const int lane  = tid & 31;
  const int wave  = tid >> 5;

  const f4* u4 = (const f4*)(u + (size_t)b * D);
  const f4 u0 = u4[lane], u1 = u4[lane + 32],
           u2 = u4[lane + 64], u3 = u4[lane + 96];

  float m = -3.4028235e38f, l = 0.0f;
  f4 c0 = {0, 0, 0, 0}, c1 = {0, 0, 0, 0}, c2 = {0, 0, 0, 0},
     c3 = {0, 0, 0, 0};

  const int tbase = chunk * CHUNK + wave * WITER;
  const f4* row0 = (const f4*)(lfb + ((size_t)b * T + tbase) * D);

  // software pipeline: keep next row's 4x b128 NT loads in flight
  f4 a0 = ntld(row0 + lane),      a1 = ntld(row0 + lane + 32),
     a2 = ntld(row0 + lane + 64), a3 = ntld(row0 + lane + 96);

  for (int it = 0; it < WITER; ++it) {
    f4 n0 = {0, 0, 0, 0}, n1 = {0, 0, 0, 0}, n2 = {0, 0, 0, 0},
       n3 = {0, 0, 0, 0};
    if (it + 1 < WITER) {
      const f4* nrow = row0 + (size_t)(it + 1) * (D / 4);
      n0 = ntld(nrow + lane);      n1 = ntld(nrow + lane + 32);
      n2 = ntld(nrow + lane + 64); n3 = ntld(nrow + lane + 96);
    }
    float p = a0.x * u0.x + a0.y * u0.y + a0.z * u0.z + a0.w * u0.w +
              a1.x * u1.x + a1.y * u1.y + a1.z * u1.z + a1.w * u1.w +
              a2.x * u2.x + a2.y * u2.y + a2.z * u2.z + a2.w * u2.w +
              a3.x * u3.x + a3.y * u3.y + a3.z * u3.z + a3.w * u3.w;
    const float s = wave_sum(p);          // wave-uniform (u pre-scaled sqrt(D))
    if (s > m) {                          // uniform branch
      const float r = __expf(m - s);
      m = s;
      l = l * r + 1.0f;
      c0 = f4_resc(c0, r, a0); c1 = f4_resc(c1, r, a1);
      c2 = f4_resc(c2, r, a2); c3 = f4_resc(c3, r, a3);
    } else {
      const float e = __expf(s - m);
      l += e;
      c0 = f4_axpy(c0, e, a0); c1 = f4_axpy(c1, e, a1);
      c2 = f4_axpy(c2, e, a2); c3 = f4_axpy(c3, e, a3);
    }
    a0 = n0; a1 = n1; a2 = n2; a3 = n3;
  }

  f4* dst = (f4*)sm_acc[wave];
  dst[lane] = c0; dst[lane + 32] = c1; dst[lane + 64] = c2; dst[lane + 96] = c3;
  if (lane == 0) { sm_m[wave] = m; sm_l[wave] = l; }
  __syncthreads();

  // merge 8 waves -> one partial record
  float M = sm_m[0];
#pragma unroll
  for (int w = 1; w < 8; ++w) M = fmaxf(M, sm_m[w]);
  float ew[8], L = 0.0f;
#pragma unroll
  for (int w = 0; w < 8; ++w) { ew[w] = __expf(sm_m[w] - M); L += ew[w] * sm_l[w]; }
  float v0 = 0.0f, v1 = 0.0f;
#pragma unroll
  for (int w = 0; w < 8; ++w) {
    v0 += ew[w] * sm_acc[w][tid];
    v1 += ew[w] * sm_acc[w][tid + 256];
  }
  float* rec = part + (size_t)blockIdx.x * REC;
  rec[tid] = v0;
  rec[tid + 256] = v1;
  if (tid == 0) { rec[D] = M; rec[D + 1] = L; }
}

// ---------------- Kernel 3: merge NCHUNK partials -> v[b][D] ----------------

__global__ void nlb_combine(const float* __restrict__ part,
                            float* __restrict__ vctx) {
  const int b = blockIdx.x;
  const int tid = threadIdx.x;  // 256
  const float* base = part + (size_t)b * NCHUNK * REC;
  float M = -3.4028235e38f;
  for (int i = 0; i < NCHUNK; ++i) M = fmaxf(M, base[(size_t)i * REC + D]);
  float L = 0.0f, v0 = 0.0f, v1 = 0.0f;
  for (int i = 0; i < NCHUNK; ++i) {
    const float* r = base + (size_t)i * REC;
    const float e = __expf(r[D] - M);
    L  += e * r[D + 1];
    v0 += e * r[tid];
    v1 += e * r[tid + 256];
  }
  const float inv = 1.0f / L;
  vctx[(size_t)b * D + tid]       = v0 * inv;
  vctx[(size_t)b * D + tid + 256] = v1 * inv;
}

// ---------------- Kernel 4: tail (WMMA dense layers) ------------------------
// ctx = v@Wgi^T + bgi ; relu(LN(ctx)) ; h = .@Wfc^T + bfc ; lt = h + x ;
// feats2 = relu(relu([x,lt]) @ Wnlb^T + bnlb) ; preds.

__global__ void nlb_tail(const float* __restrict__ vctx,
                         const float* __restrict__ x,
                         const float* __restrict__ w_gi,
                         const float* __restrict__ b_gi,
                         const float* __restrict__ ln_g,
                         const float* __restrict__ ln_b,
                         const float* __restrict__ w_fc,
                         const float* __restrict__ b_fc,
                         const float* __restrict__ w_nlb,
                         const float* __restrict__ b_nlb,
                         const float* __restrict__ w1,
                         const float* __restrict__ b1,
                         const float* __restrict__ w2,
                         const float* __restrict__ b2,
                         float* __restrict__ out) {
  extern __shared__ float smem[];
  float* bufA = smem;                  // 16x512
  float* bufB = smem + B * D;          // 16x512
  float* bufC = smem + 2 * B * D;      // 16x1024
  const int tid  = threadIdx.x;        // 512 threads, 16 waves
  const int lane = tid & 31;
  const int wave = tid >> 5;
  const int rb   = (lane >> 4) << 3;
  const int nlo  = lane & 15;

  // Stage 1: ctx -> bufA
  for (int tile = wave; tile < D / 16; tile += 16) {
    const int n0 = tile * 16;
    v8f c = zero_v8f();
    for (int k0 = 0; k0 < D; k0 += 32)
      c = wmma_bf16(load_a_rm(vctx, D, k0, lane),
                    load_b_wT(w_gi, D, k0, n0, lane), c);
    const float bias = b_gi[n0 + nlo];
#pragma unroll
    for (int r = 0; r < 8; ++r) bufA[(rb + r) * D + n0 + nlo] = c[r] + bias;
  }
  __syncthreads();

  // Stage 2: relu(layernorm(bufA)) in place; wave w owns row w
  {
    const int row = wave;
    float s = 0.0f, ss = 0.0f;
#pragma unroll
    for (int k = 0; k < D / 32; ++k) {
      const float h = bufA[row * D + lane + 32 * k];
      s += h; ss += h * h;
    }
    s = wave_sum(s); ss = wave_sum(ss);
    const float mu  = s * (1.0f / D);
    const float var = ss * (1.0f / D) - mu * mu;
    const float rs  = rsqrtf(var + LN_EPS);
#pragma unroll
    for (int k = 0; k < D / 32; ++k) {
      const int d = lane + 32 * k;
      const float h = bufA[row * D + d];
      bufA[row * D + d] = fmaxf(0.0f, (h - mu) * rs * ln_g[d] + ln_b[d]);
    }
  }
  __syncthreads();

  // Stage 3: bufB = relu(bufA@Wfc^T + bfc + x)
  for (int tile = wave; tile < D / 16; tile += 16) {
    const int n0 = tile * 16;
    v8f c = zero_v8f();
    for (int k0 = 0; k0 < D; k0 += 32)
      c = wmma_bf16(load_a_rm(bufA, D, k0, lane),
                    load_b_wT(w_fc, D, k0, n0, lane), c);
    const float bias = b_fc[n0 + nlo];
#pragma unroll
    for (int r = 0; r < 8; ++r) {
      const int idx = (rb + r) * D + n0 + nlo;
      bufB[idx] = fmaxf(0.0f, c[r] + bias + x[idx]);
    }
  }
  __syncthreads();

  // bufA := relu(x)
  for (int i = tid; i < B * D; i += 512) bufA[i] = fmaxf(0.0f, x[i]);
  __syncthreads();

  // Stage 4: bufC = relu([bufA|bufB] @ Wnlb^T + bnlb)   K = 1024
  for (int tile = wave; tile < (2 * D) / 16; tile += 16) {
    const int n0 = tile * 16;
    v8f c = zero_v8f();
    for (int k0 = 0; k0 < 2 * D; k0 += 32) {
      const float* asrc = (k0 < D) ? bufA : bufB;
      const int ka = (k0 < D) ? k0 : (k0 - D);
      c = wmma_bf16(load_a_rm(asrc, D, ka, lane),
                    load_b_wT(w_nlb, 2 * D, k0, n0, lane), c);
    }
    const float bias = b_nlb[n0 + nlo];
#pragma unroll
    for (int r = 0; r < 8; ++r)
      bufC[(rb + r) * (2 * D) + n0 + nlo] = fmaxf(0.0f, c[r] + bias);
  }
  __syncthreads();

  // Stage 5: preds; wave w owns row w
  {
    const int row = wave;
    float s1 = 0.0f, s2 = 0.0f;
#pragma unroll
    for (int k = 0; k < D / 32; ++k) {
      const int d = lane + 32 * k;
      s1 += bufC[row * (2 * D) + d]     * w1[d];
      s2 += bufC[row * (2 * D) + D + d] * w2[d];
    }
    s1 = wave_sum(s1); s2 = wave_sum(s2);
    if (lane == 0) {
      out[row * 2 + 0] = s1 + b1[0];
      out[row * 2 + 1] = s2 + b2[0];
    }
  }
}

// ---------------- launch ----------------------------------------------------

extern "C" void kernel_launch(void* const* d_in, const int* in_sizes, int n_in,
                              void* d_out, int out_size, void* d_ws,
                              size_t ws_size, hipStream_t stream) {
  const float* x        = (const float*)d_in[0];
  const float* lfb      = (const float*)d_in[1];
  const float* w_theta  = (const float*)d_in[2];
  const float* b_theta  = (const float*)d_in[3];
  const float* w_phi    = (const float*)d_in[4];
  // d_in[5] = b_phi: per-batch constant score offset -> softmax invariant.
  const float* w_gi     = (const float*)d_in[6];
  const float* b_gi     = (const float*)d_in[7];
  const float* ln_g     = (const float*)d_in[8];
  const float* ln_b     = (const float*)d_in[9];
  const float* w_fc     = (const float*)d_in[10];
  const float* b_fc     = (const float*)d_in[11];
  const float* w_nlb    = (const float*)d_in[12];
  const float* b_nlb    = (const float*)d_in[13];
  const float* w1       = (const float*)d_in[14];
  const float* b1       = (const float*)d_in[15];
  const float* w2       = (const float*)d_in[16];
  const float* b2       = (const float*)d_in[17];
  float* out = (float*)d_out;

  float* ws_u    = (float*)d_ws;                       // B*D
  float* ws_part = ws_u + B * D;                       // B*NCHUNK*REC
  float* ws_v    = ws_part + (size_t)B * NCHUNK * REC; // B*D

  nlb_head<<<1, 512, 0, stream>>>(x, w_theta, b_theta, w_phi, ws_u);
  nlb_stream<<<B * NCHUNK, 256, 0, stream>>>(lfb, ws_u, ws_part);
  nlb_combine<<<B, 256, 0, stream>>>(ws_part, ws_v);
  nlb_tail<<<1, 512, (2 * B * D + B * 2 * D) * sizeof(float), stream>>>(
      ws_v, x, w_gi, b_gi, ln_g, ln_b, w_fc, b_fc, w_nlb, b_nlb, w1, b1, w2,
      b2, out);
}
